// SE3Transformer_15126874816932
// MI455X (gfx1250) — matrix operand order repaired and checked by reading, not compiled, not run on hardware
//
#include <hip/hip_runtime.h>

// -------- CDNA5 WMMA types --------
typedef _Float16 v16h __attribute__((ext_vector_type(16)));
typedef _Float16 v8h  __attribute__((ext_vector_type(8)));
typedef float    v8f  __attribute__((ext_vector_type(8)));

#define N_NODE   16384
#define N_EDGE   262144

// ---- constants from the reference ----
#define C_INV3    0.57735026919f       // 1/sqrt(3)
#define C_SQRT3   1.73205080757f
#define C_RS20    0.22360679775f       // 1/sqrt(20)
#define C_RS16    0.25f                // 1/sqrt(16)
#define C_RS8     0.35355339059f       // 1/sqrt(8)
#define C_NORM0   0.04564354646f       // RS20 * 1/sqrt(24) (mlp fan-in + TP norm)
#define C_NORM1   0.02635231383f       // C_NORM0 * INV3
#define C_FAN     0.11180339887f       // 1/sqrt(8*8+4*4)
#define C_STEP    0.47619047619f       // 10/21
#define C_INVSTEP 2.1f
#define C_EMBC    37.716084f           // 1.14136*e^2*sqrt(20)
#define C_EPS     1e-6f

// packed-f16 fragment array sizes (halves)
#define SZ_W1T   1024                  // 32 cols x 32 K
#define SZ_WK2T  9216                  // 288 cols x 32 K
#define SZ_WV2T  18432                 // 576 cols x 32 K

__device__ __forceinline__ float sus(float x) {          // soft_unit_step
  return (x > 0.f) ? __expf(-1.f / x) : 0.f;
}

// A fragment: 16 halves for this lane from a [row][32] K-major buffer.
// lanes 0-15: K = {0..15}; lanes 16-31: K = {16..31}.
__device__ __forceinline__ v16h load_afrag(const _Float16* rowp, int kb2) {
  v8h lo = *(const v8h*)(rowp + 2 * kb2);
  v8h hi = *(const v8h*)(rowp + 2 * kb2 + 8);
  return __builtin_shufflevector(lo, hi, 0, 1, 2, 3, 4, 5, 6, 7,
                                 8, 9, 10, 11, 12, 13, 14, 15);
}

// ---------------- per-wave LDS tile state ----------------
struct __align__(16) WaveLds {
  int      srcv[16];
  float    vlen[16];
  float    sh1[16][3];     // sqrt(3)*unit vec
  float    f0s[16][16];    // f0[src]
  float    f1s[16][24];    // f1[src] flattened (i*3+c)
  float    dd[16][8];      // INV3 * <f1, sh1>
  _Float16 embbuf[16][32]; // radial embedding, K-padded
  _Float16 hbuf[16][32];   // relu(emb@W1)/sqrt(20), K-padded
  float    acck[16][20];   // k0[8], k1[4][3] raw accumulators
  float    accv[16][40];   // v0[16], v1[8][3] raw accumulators
};

// one 16-col tile: B frag load (32B) + WMMA
__device__ __forceinline__ v8f tile_mm(const v16h& a, const v16h* bp, int idx) {
  v16h b = bp[idx];
  v8f c = {};
  return __builtin_amdgcn_wmma_f32_16x16x32_f16(false, a, false, b, (short)0, c,
                                                false, false);
}

// emb(A frag) @ W1t -> hbuf (relu + 1/sqrt(20))
__device__ __forceinline__ void hidden_gemm(const v16h& a, const _Float16* __restrict__ w1t,
                                            _Float16 (*hbuf)[32], int lane) {
  const int hi01 = (lane & 16) ? 1 : 0;
  const int cl   = lane & 15;
  const int mrow = (lane & 16) ? 8 : 0;
  const v16h* bp = (const v16h*)w1t;
#pragma unroll
  for (int t = 0; t < 2; ++t) {
    int col = t * 16 + cl;
    v8f c = tile_mm(a, bp, col * 2 + hi01);
#pragma unroll
    for (int r = 0; r < 8; ++r) {
      float h = fmaxf(c[r] * C_RS20, 0.f);
      hbuf[r + mrow][col] = (_Float16)h;
    }
  }
}

// ========== prep: pack weights into fragment-ready f16 W^T[col][K(32)] ======
__global__ __launch_bounds__(256) void k_prep(const float* __restrict__ Wk1,
    const float* __restrict__ Wv1, const float* __restrict__ Wk2,
    const float* __restrict__ Wv2, _Float16* __restrict__ wk1t,
    _Float16* __restrict__ wv1t, _Float16* __restrict__ wk2t,
    _Float16* __restrict__ wv2t) {
  int id = blockIdx.x * 256 + threadIdx.x;
  if (id < SZ_W1T) {
    int col = id >> 5, K = id & 31;
    wk1t[id] = (_Float16)((col < 20 && K < 20) ? Wk1[K * 20 + col] : 0.f);
  } else if (id < 2 * SZ_W1T) {
    int j = id - SZ_W1T, col = j >> 5, K = j & 31;
    wv1t[j] = (_Float16)((col < 20 && K < 20) ? Wv1[K * 20 + col] : 0.f);
  } else if (id < 2 * SZ_W1T + SZ_WK2T) {
    int j = id - 2 * SZ_W1T, col = j >> 5, K = j & 31;
    wk2t[j] = (_Float16)((K < 20) ? Wk2[K * 288 + col] : 0.f);
  } else if (id < 2 * SZ_W1T + SZ_WK2T + SZ_WV2T) {
    int j = id - (2 * SZ_W1T + SZ_WK2T), col = j >> 5, K = j & 31;
    wv2t[j] = (_Float16)((K < 20) ? Wv2[K * 576 + col] : 0.f);
  }
}

// ================= K0: node init (q, self-interaction, z=0) =================
__global__ __launch_bounds__(256) void k_node(const float* __restrict__ f_in,
    const float* __restrict__ Wq0, const float* __restrict__ Wq1,
    const float* __restrict__ Ws0, const float* __restrict__ Ws1,
    float* __restrict__ qbuf, float* __restrict__ z, float* __restrict__ out) {
  int n = blockIdx.x * blockDim.x + threadIdx.x;
  if (n >= N_NODE) return;
  float f0[16], f1[24];
#pragma unroll
  for (int i = 0; i < 16; ++i) f0[i] = f_in[n * 40 + i];
#pragma unroll
  for (int j = 0; j < 24; ++j) f1[j] = f_in[n * 40 + 16 + j];
  for (int o = 0; o < 8; ++o) {
    float s = 0.f;
#pragma unroll
    for (int i = 0; i < 16; ++i) s += f0[i] * Wq0[i * 8 + o];
    qbuf[n * 20 + o] = s * C_RS16;
  }
  for (int o = 0; o < 4; ++o)
    for (int c = 0; c < 3; ++c) {
      float s = 0.f;
#pragma unroll
      for (int i = 0; i < 8; ++i) s += f1[i * 3 + c] * Wq1[i * 4 + o];
      qbuf[n * 20 + 8 + o * 3 + c] = s * C_RS8;
    }
  for (int o = 0; o < 16; ++o) {
    float s = 0.f;
#pragma unroll
    for (int i = 0; i < 16; ++i) s += f0[i] * Ws0[i * 16 + o];
    out[n * 40 + o] = s * C_RS16;
  }
  for (int o = 0; o < 8; ++o)
    for (int c = 0; c < 3; ++c) {
      float s = 0.f;
#pragma unroll
      for (int i = 0; i < 8; ++i) s += f1[i * 3 + c] * Ws1[i * 8 + o];
      out[n * 40 + 16 + o * 3 + c] = s * C_RS8;
    }
  z[n] = 0.f;
}

// ================= K1: fused edge kernel (WMMA MLPs + tensor products) ======
__global__ __launch_bounds__(64) void k_edge(const float* __restrict__ pos,
    const float* __restrict__ f_in,
    const int* __restrict__ esrc, const int* __restrict__ edst,
    const _Float16* __restrict__ wk1t, const _Float16* __restrict__ wk2t,
    const _Float16* __restrict__ wv1t, const _Float16* __restrict__ wv2t,
    const float* __restrict__ Wd0, const float* __restrict__ Wd1,
    const float* __restrict__ qbuf,
    float* __restrict__ logitbuf, float* __restrict__ cutbuf,
    float* __restrict__ vbuf) {
  __shared__ WaveLds smem[2];
  const int lane = threadIdx.x & 31;
  const int wid  = threadIdx.x >> 5;
  WaveLds& L = smem[wid];
  const int tile  = blockIdx.x * 2 + wid;
  const int ebase = tile * 16;

  // ---- setup: geometry (lanes 0-15, one edge each) ----
  int dn = 0;
  if (lane < 16) {
    int ge = ebase + lane;
    int s = esrc[ge];
    dn    = edst[ge];
    L.srcv[lane] = s;
    float dx = pos[dn * 3 + 0] - pos[s * 3 + 0];
    float dy = pos[dn * 3 + 1] - pos[s * 3 + 1];
    float dz = pos[dn * 3 + 2] - pos[s * 3 + 2];
    float len = sqrtf(dx * dx + dy * dy + dz * dz);
    float inv = C_SQRT3 / fmaxf(len, 1e-9f);
    L.sh1[lane][0] = dx * inv;
    L.sh1[lane][1] = dy * inv;
    L.sh1[lane][2] = dz * inv;
    L.vlen[lane] = len;
    cutbuf[ge] = sus(10.f - len);      // soft_unit_step(10*(1 - len/R))
  }
  __syncthreads();

  // ---- gather source features (L2-resident), zero accumulators ----
  for (int t = lane; t < 640; t += 32) {
    int e = t / 40, j = t % 40;
    float val = f_in[L.srcv[e] * 40 + j];
    if (j < 16) L.f0s[e][j] = val;
    else        L.f1s[e][j - 16] = val;
  }
  for (int t = lane; t < 320; t += 32) ((float*)L.acck)[t] = 0.f;
  for (int t = lane; t < 640; t += 32) ((float*)L.accv)[t] = 0.f;
  // ---- radial embedding into K-major f16 buffer (cooperative) ----
  for (int t = lane; t < 320; t += 32) {
    int e = t / 20, j = t % 20;
    float x = (L.vlen[e] - (j + 1) * C_STEP) * C_INVSTEP;
    L.embbuf[e][j] = (_Float16)(C_EMBC * sus(x + 1.f) * sus(1.f - x));
  }
  for (int t = lane; t < 192; t += 32) {
    int e = t / 12, j = 20 + t % 12;
    L.embbuf[e][j] = (_Float16)0.f;
  }
  __syncthreads();

  // ---- d[e,i] = INV3 * <f1[e,i,:], sh1[e,:]> ----
  for (int t = lane; t < 128; t += 32) {
    int e = t >> 3, i = t & 7;
    L.dd[e][i] = C_INV3 * (L.f1s[e][i * 3 + 0] * L.sh1[e][0] +
                           L.f1s[e][i * 3 + 1] * L.sh1[e][1] +
                           L.f1s[e][i * 3 + 2] * L.sh1[e][2]);
  }
  __syncthreads();

  const int erow = lane & 15;
  const int cl   = lane & 15;
  const int hi01 = (lane & 16) ? 1 : 0;
  const int kb2  = (lane & 16) ? 8 : 0;
  const int mrow = (lane & 16) ? 8 : 0;
  // hoisted per-lane (i,o) decode pieces (tile-block boundaries are multiples
  // of 16 columns, so the path choice is uniform per tile -> scalar branches)
  const int cl3 = cl >> 3, cl7 = cl & 7;
  const int cl4 = cl >> 4;               // 0 for 16-wide: unused pieces folded
  (void)cl4;
  const int cl2 = cl >> 2, clo3 = cl & 3;

  // A operand: radial embedding fragment (2x ds_load_b128)
  v16h a = load_afrag(&L.embbuf[erow][0], kb2);

  // ======================== K path ========================
  hidden_gemm(a, wk1t, L.hbuf, lane);
  __syncthreads();
  v16h ah = load_afrag(&L.hbuf[erow][0], kb2);

  const v16h* bk2 = (const v16h*)wk2t;
  // w0a: cols [0,128): i = 2t + cl>>3, o = cl&7
#pragma unroll 2
  for (int t = 0; t < 8; ++t) {
    v8f c = tile_mm(ah, bk2, (t * 16 + cl) * 2 + hi01);
    int i = 2 * t + cl3, o = cl7;
#pragma unroll
    for (int r = 0; r < 8; ++r) {
      int e = r + mrow;
      atomicAdd(&L.acck[e][o], c[r] * L.f0s[e][i]);
    }
  }
  // w0b: cols [128,192): i = 2(t-8) + cl>>3, o = cl&7
#pragma unroll 2
  for (int t = 8; t < 12; ++t) {
    v8f c = tile_mm(ah, bk2, (t * 16 + cl) * 2 + hi01);
    int i = 2 * (t - 8) + cl3, o = cl7;
#pragma unroll
    for (int r = 0; r < 8; ++r) {
      int e = r + mrow;
      atomicAdd(&L.acck[e][o], c[r] * L.dd[e][i]);
    }
  }
  // w1a: cols [192,256): i = 4(t-12) + cl>>2, o = cl&3
#pragma unroll 2
  for (int t = 12; t < 16; ++t) {
    v8f c = tile_mm(ah, bk2, (t * 16 + cl) * 2 + hi01);
    int i = 4 * (t - 12) + cl2, o = clo3;
#pragma unroll
    for (int r = 0; r < 8; ++r) {
      int e = r + mrow;
      float fw = c[r] * L.f0s[e][i];
      atomicAdd(&L.acck[e][8 + o * 3 + 0], fw * L.sh1[e][0]);
      atomicAdd(&L.acck[e][8 + o * 3 + 1], fw * L.sh1[e][1]);
      atomicAdd(&L.acck[e][8 + o * 3 + 2], fw * L.sh1[e][2]);
    }
  }
  // w1b: cols [256,288): i = 4(t-16) + cl>>2, o = cl&3
#pragma unroll 2
  for (int t = 16; t < 18; ++t) {
    v8f c = tile_mm(ah, bk2, (t * 16 + cl) * 2 + hi01);
    int i = 4 * (t - 16) + cl2, o = clo3;
#pragma unroll
    for (int r = 0; r < 8; ++r) {
      int e = r + mrow;
      atomicAdd(&L.acck[e][8 + o * 3 + 0], c[r] * L.f1s[e][i * 3 + 0]);
      atomicAdd(&L.acck[e][8 + o * 3 + 1], c[r] * L.f1s[e][i * 3 + 1]);
      atomicAdd(&L.acck[e][8 + o * 3 + 2], c[r] * L.f1s[e][i * 3 + 2]);
    }
  }
  __syncthreads();

  // ---- attention logit (lanes 0-15, one edge each) ----
  if (lane < 16) {
    int e = lane, ge = ebase + e;
    float k0f[8];
#pragma unroll
    for (int j = 0; j < 8; ++j) k0f[j] = L.acck[e][j] * C_NORM0;
    const float* qb = qbuf + dn * 20;
    float s0 = 0.f;
    for (int j = 0; j < 8; ++j) {
      float tj = 0.f;
#pragma unroll
      for (int i = 0; i < 8; ++i) tj += qb[i] * Wd0[i * 8 + j];
      s0 += tj * k0f[j];
    }
    float s1 = 0.f;
    for (int j = 0; j < 4; ++j)
      for (int c = 0; c < 3; ++c) {
        float u = 0.f;
#pragma unroll
        for (int i = 0; i < 4; ++i) u += qb[8 + i * 3 + c] * Wd1[i * 4 + j];
        s1 += u * (L.acck[e][8 + j * 3 + c] * C_NORM1);
      }
    logitbuf[ge] = (s0 + C_INV3 * s1) * C_FAN;
  }
  __syncthreads();

  // ======================== V path ========================
  hidden_gemm(a, wv1t, L.hbuf, lane);
  __syncthreads();
  ah = load_afrag(&L.hbuf[erow][0], kb2);

  const v16h* bv2 = (const v16h*)wv2t;
  // w0a: cols [0,256): i = t, o = cl
#pragma unroll 2
  for (int t = 0; t < 16; ++t) {
    v8f c = tile_mm(ah, bv2, (t * 16 + cl) * 2 + hi01);
    int i = t, o = cl;
#pragma unroll
    for (int r = 0; r < 8; ++r) {
      int e = r + mrow;
      atomicAdd(&L.accv[e][o], c[r] * L.f0s[e][i]);
    }
  }
  // w0b: cols [256,384): i = t-16, o = cl
#pragma unroll 2
  for (int t = 16; t < 24; ++t) {
    v8f c = tile_mm(ah, bv2, (t * 16 + cl) * 2 + hi01);
    int i = t - 16, o = cl;
#pragma unroll
    for (int r = 0; r < 8; ++r) {
      int e = r + mrow;
      atomicAdd(&L.accv[e][o], c[r] * L.dd[e][i]);
    }
  }
  // w1a: cols [384,512): i = 2(t-24) + cl>>3, o = cl&7
#pragma unroll 2
  for (int t = 24; t < 32; ++t) {
    v8f c = tile_mm(ah, bv2, (t * 16 + cl) * 2 + hi01);
    int i = 2 * (t - 24) + cl3, o = cl7;
#pragma unroll
    for (int r = 0; r < 8; ++r) {
      int e = r + mrow;
      float fw = c[r] * L.f0s[e][i];
      atomicAdd(&L.accv[e][16 + o * 3 + 0], fw * L.sh1[e][0]);
      atomicAdd(&L.accv[e][16 + o * 3 + 1], fw * L.sh1[e][1]);
      atomicAdd(&L.accv[e][16 + o * 3 + 2], fw * L.sh1[e][2]);
    }
  }
  // w1b: cols [512,576): i = 2(t-32) + cl>>3, o = cl&7
#pragma unroll 2
  for (int t = 32; t < 36; ++t) {
    v8f c = tile_mm(ah, bv2, (t * 16 + cl) * 2 + hi01);
    int i = 2 * (t - 32) + cl3, o = cl7;
#pragma unroll
    for (int r = 0; r < 8; ++r) {
      int e = r + mrow;
      atomicAdd(&L.accv[e][16 + o * 3 + 0], c[r] * L.f1s[e][i * 3 + 0]);
      atomicAdd(&L.accv[e][16 + o * 3 + 1], c[r] * L.f1s[e][i * 3 + 1]);
      atomicAdd(&L.accv[e][16 + o * 3 + 2], c[r] * L.f1s[e][i * 3 + 2]);
    }
  }
  __syncthreads();

  // ---- write per-edge v (40 floats) ----
  for (int t = lane; t < 640; t += 32) {
    int e = t / 40, j = t % 40;
    float sc = (j < 16) ? C_NORM0 : C_NORM1;
    vbuf[(ebase + e) * 40 + j] = L.accv[e][j] * sc;
  }
}

// ================= K2: global max of logits (2 stages) ======================
__global__ __launch_bounds__(256) void k_max1(const float* __restrict__ logit,
                                              float* __restrict__ partial) {
  __shared__ float red[256];
  int tid = threadIdx.x;
  float m = -1e30f;
  for (int i = blockIdx.x * 256 + tid; i < N_EDGE; i += 256 * 256)
    m = fmaxf(m, logit[i]);
  red[tid] = m;
  __syncthreads();
  for (int s = 128; s > 0; s >>= 1) {
    if (tid < s) red[tid] = fmaxf(red[tid], red[tid + s]);
    __syncthreads();
  }
  if (tid == 0) partial[blockIdx.x] = red[0];
}

__global__ __launch_bounds__(256) void k_max2(const float* __restrict__ partial,
                                              float* __restrict__ gmax) {
  __shared__ float red[256];
  int tid = threadIdx.x;
  red[tid] = partial[tid];
  __syncthreads();
  for (int s = 128; s > 0; s >>= 1) {
    if (tid < s) red[tid] = fmaxf(red[tid], red[tid + s]);
    __syncthreads();
  }
  if (tid == 0) gmax[0] = red[0];
}

// ================= K3: exp + segment-sum z ==================================
__global__ __launch_bounds__(256) void k_exp(const float* __restrict__ logit,
    const float* __restrict__ cut, const float* __restrict__ gmax,
    const int* __restrict__ edst, float* __restrict__ exb, float* __restrict__ z) {
  int e = blockIdx.x * 256 + threadIdx.x;
  if (e >= N_EDGE) return;
  float ex = cut[e] * __expf(logit[e] - gmax[0]);
  exb[e] = ex;
  atomicAdd(&z[edst[e]], ex);
}

// ================= K4: normalized scatter of w*v ============================
__global__ __launch_bounds__(256) void k_scatter(const float* __restrict__ exb,
    const float* __restrict__ z, const int* __restrict__ edst,
    const float* __restrict__ vbuf, float* __restrict__ out) {
  int g = blockIdx.x * 256 + threadIdx.x;
  if (g >= N_EDGE * 40) return;
  int e = g / 40;
  int dnode = edst[e];
  float zz = z[dnode];
  zz = (zz < C_EPS) ? 1.f : zz;
  float alpha = exb[e] / zz;
  float w = sqrtf(fmaxf(alpha, 0.f) + C_EPS);
  atomicAdd(&out[dnode * 40 + (g % 40)], w * vbuf[g]);
}

// ============================================================================
extern "C" void kernel_launch(void* const* d_in, const int* in_sizes, int n_in,
                              void* d_out, int out_size, void* d_ws, size_t ws_size,
                              hipStream_t stream) {
  const float* pos  = (const float*)d_in[0];
  const float* f_in = (const float*)d_in[1];
  const int*   esrc = (const int*)d_in[2];
  const int*   edst = (const int*)d_in[3];
  const float* Wq0  = (const float*)d_in[4];
  const float* Wq1  = (const float*)d_in[5];
  const float* Wk1  = (const float*)d_in[6];
  const float* Wk2  = (const float*)d_in[7];
  const float* Wv1  = (const float*)d_in[8];
  const float* Wv2  = (const float*)d_in[9];
  const float* Wd0  = (const float*)d_in[10];
  const float* Wd1  = (const float*)d_in[11];
  const float* Ws0  = (const float*)d_in[12];
  const float* Ws1  = (const float*)d_in[13];
  float* out = (float*)d_out;

  // workspace partition (~46.6 MB; all offsets 64B-aligned)
  float* w = (float*)d_ws;
  float* qbuf    = w; w += N_NODE * 20;
  float* logitb  = w; w += N_EDGE;
  float* cutb    = w; w += N_EDGE;
  float* exb     = w; w += N_EDGE;
  float* zbuf    = w; w += N_NODE;
  float* partial = w; w += 256;
  float* gmax    = w; w += 16;
  float* vbuf    = w; w += (size_t)N_EDGE * 40;
  _Float16* hw = (_Float16*)w;
  _Float16* wk1t = hw;                        // 1024 halves
  _Float16* wv1t = hw + SZ_W1T;               // 1024
  _Float16* wk2t = hw + 2 * SZ_W1T;           // 9216
  _Float16* wv2t = hw + 2 * SZ_W1T + SZ_WK2T; // 18432

  k_prep<<<(2 * SZ_W1T + SZ_WK2T + SZ_WV2T + 255) / 256, 256, 0, stream>>>(
      Wk1, Wv1, Wk2, Wv2, wk1t, wv1t, wk2t, wv2t);
  k_node<<<N_NODE / 256, 256, 0, stream>>>(f_in, Wq0, Wq1, Ws0, Ws1, qbuf, zbuf, out);
  k_edge<<<N_EDGE / 32, 64, 0, stream>>>(pos, f_in, esrc, edst, wk1t, wk2t, wv1t, wv2t,
                                         Wd0, Wd1, qbuf, logitb, cutb, vbuf);
  k_max1<<<256, 256, 0, stream>>>(logitb, partial);
  k_max2<<<1, 256, 0, stream>>>(partial, gmax);
  k_exp<<<N_EDGE / 256, 256, 0, stream>>>(logitb, cutb, gmax, edst, exb, zbuf);
  k_scatter<<<(N_EDGE * 40) / 256, 256, 0, stream>>>(exb, zbuf, edst, vbuf, out);
}
